// AttentionBlock_9706626089938
// MI455X (gfx1250) — matrix operand order repaired and checked by reading, not compiled, not run on hardware
//
#include <hip/hip_runtime.h>
#include <hip/hip_bf16.h>

// ---------------------------------------------------------------------------
// AttentionBlock for MI455X (gfx1250): GroupNorm -> QKV gemm -> flash attn ->
// proj gemm + residual. All matmuls use v_wmma_f32_16x16x32_bf16 (bf16 inputs,
// f32 accumulate). Fragment loads are 2x ds_load_b128 each. Attention runs
// 4 waves / WG (1 wave per SIMD) so Q fragments + O accumulators stay fully
// register-resident (no scratch spills).
// b=4, c=512, h=w=64, N=4096, groups=32.
// ---------------------------------------------------------------------------

typedef __attribute__((ext_vector_type(16))) __bf16 v16bf;
typedef __attribute__((ext_vector_type(8)))  float  v8f;

#define CDIM 512
#define NSP  4096
#define BATCH 4

union Frag16 {           // one WMMA 16-bf16 operand = two 16B LDS loads
    v16bf v;
    uint4 q[2];
    unsigned short u[16];
};
union Pack8 {            // 8 ushorts <-> uint4
    uint4 q;
    unsigned short u[8];
};

__device__ __forceinline__ unsigned short f2bf(float f) {
    unsigned u = __builtin_bit_cast(unsigned, f);
    u += 0x7FFFu + ((u >> 16) & 1u);           // round-to-nearest-even
    return (unsigned short)(u >> 16);
}

// A-fragment (16x32, M across lanes): lane ll = row; runs K=8h..+7, 16+8h..+7
__device__ __forceinline__ v16bf load_afrag(const unsigned short* row, int lh) {
    Frag16 f;
    f.q[0] = *(const uint4*)(row + 8 * lh);
    f.q[1] = *(const uint4*)(row + 16 + 8 * lh);
    return f.v;
}
// B-fragment (32x16, N across lanes) from a TRANSPOSED tile row [n][k]:
// runs K=8h..+7, 16+8h..+7
__device__ __forceinline__ v16bf load_bfrag(const unsigned short* row, int lh) {
    Frag16 f;
    f.q[0] = *(const uint4*)(row + 8 * lh);
    f.q[1] = *(const uint4*)(row + 16 + 8 * lh);
    return f.v;
}

// ---------------------------------------------------------------------------
// One-shot fp32 -> bf16 weight conversion (bandwidth-trivial).
// ---------------------------------------------------------------------------
__global__ __launch_bounds__(256)
void wconv_kernel(const float* __restrict__ w, unsigned short* __restrict__ o, int n4)
{
    const int i = blockIdx.x * 256 + threadIdx.x;
    if (i < n4) {
        float4 v4 = ((const float4*)w)[i];
        uint2 p;
        p.x = (unsigned)f2bf(v4.x) | ((unsigned)f2bf(v4.y) << 16);
        p.y = (unsigned)f2bf(v4.z) | ((unsigned)f2bf(v4.w) << 16);
        ((uint2*)o)[i] = p;
    }
}

// ---------------------------------------------------------------------------
// GroupNorm: one block per (batch, group). group = 16 ch x 4096 = 65536 elems.
// Writes bf16 normalized activations to workspace, layout [b][c][n].
// ---------------------------------------------------------------------------
__global__ __launch_bounds__(256)
void gn_kernel(const float* __restrict__ x, const float* __restrict__ gamma,
               const float* __restrict__ beta, unsigned short* __restrict__ xn)
{
    const int bg = blockIdx.x;
    const int b  = bg >> 5;
    const int g  = bg & 31;
    const size_t base = ((size_t)b * CDIM + (size_t)g * 16) * NSP;
    const float* xp = x + base;

    float sum = 0.f, sq = 0.f;
    for (int i = threadIdx.x; i < 65536 / 4; i += 256) {
        float4 v4 = ((const float4*)xp)[i];
        sum += v4.x + v4.y + v4.z + v4.w;
        sq  += v4.x * v4.x + v4.y * v4.y + v4.z * v4.z + v4.w * v4.w;
    }
    #pragma unroll
    for (int m = 1; m < 32; m <<= 1) {
        sum += __shfl_xor(sum, m, 32);
        sq  += __shfl_xor(sq,  m, 32);
    }
    __shared__ float ssum[8], ssq[8];
    if ((threadIdx.x & 31) == 0) {
        ssum[threadIdx.x >> 5] = sum;
        ssq [threadIdx.x >> 5] = sq;
    }
    __syncthreads();
    float ts = 0.f, tq = 0.f;
    #pragma unroll
    for (int wv = 0; wv < 8; ++wv) { ts += ssum[wv]; tq += ssq[wv]; }
    const float mean = ts * (1.0f / 65536.0f);
    const float var  = tq * (1.0f / 65536.0f) - mean * mean;
    const float inv  = rsqrtf(var + 1e-6f);

    // pass 2: float4 in, 4 packed bf16 (uint2) out
    uint2* outv = (uint2*)(xn + base);
    for (int i = threadIdx.x; i < 65536 / 4; i += 256) {
        const int ch = g * 16 + (i >> 10);          // 1024 float4s per channel
        const float gm = gamma[ch], bt = beta[ch];
        float4 v4 = ((const float4*)xp)[i];
        const unsigned b0 = f2bf((v4.x - mean) * inv * gm + bt);
        const unsigned b1 = f2bf((v4.y - mean) * inv * gm + bt);
        const unsigned b2 = f2bf((v4.z - mean) * inv * gm + bt);
        const unsigned b3 = f2bf((v4.w - mean) * inv * gm + bt);
        uint2 o; o.x = b0 | (b1 << 16); o.y = b2 | (b3 << 16);
        outv[i] = o;
    }
}

// ---------------------------------------------------------------------------
// GEMM: D[o][n] = sum_c A[o][c] * B[b][c][n]  (+ epilogue)
// MODE 0 (qkv):  A = w_qkv bf16 (1536x512), B = xn bf16; out bf16 + bias,
//                q/k rows scaled by c^-0.25.
// MODE 1 (proj): A = w_proj bf16 (512x512),  B = attn-out bf16; out f32 =
//                acc + bias + residual(x).
// WG tile 128(M) x 128(N), 8 waves each 64x32, K streamed in 32-chunks.
// ---------------------------------------------------------------------------
template <int MODE>
__global__ __launch_bounds__(256)
void gemm_kernel(const unsigned short* __restrict__ A,
                 const unsigned short* __restrict__ B,
                 const float* __restrict__ bias,
                 unsigned short* __restrict__ outb,
                 float* __restrict__ outf,
                 const float* __restrict__ resid,
                 int M, float qkscale)
{
    const int b     = blockIdx.z;
    const int nBase = blockIdx.x * 128;
    const int oBase = blockIdx.y * 128;
    const int tid  = threadIdx.x;
    const int wave = tid >> 5;
    const int lane = tid & 31;
    const int lh   = lane >> 4;    // half (0/1)
    const int ll   = lane & 15;
    const int wm   = wave >> 2;    // 0..1
    const int wn   = wave & 3;     // 0..3

    __shared__ alignas(16) unsigned short As [128][40];  // [o][c] row-major
    __shared__ alignas(16) unsigned short BsT[128][40];  // [n][c] transposed

    const size_t Bbase = (size_t)b * CDIM * NSP;

    // cooperative-load coordinates
    const int ar = tid >> 1,  ac = (tid & 1) * 16;   // A: 128 rows x 32 cols
    const int br = tid >> 3,  bc = (tid & 7) * 16;   // B: 32 rows x 128 cols

    v8f acc[8];
    #pragma unroll
    for (int t = 0; t < 8; ++t)
        #pragma unroll
        for (int r = 0; r < 8; ++r) acc[t][r] = 0.f;

    #pragma unroll
    for (int kc = 0; kc < 16; ++kc) {
        // ---- A tile: 16 contiguous bf16 per thread, pure b128 copy
        {
            const unsigned short* apb = &A[(size_t)(oBase + ar) * CDIM + kc * 32 + ac];
            uint4 a0 = *(const uint4*)(apb);
            uint4 a1 = *(const uint4*)(apb + 8);
            *(uint4*)&As[ar][ac]     = a0;
            *(uint4*)&As[ar][ac + 8] = a1;
        }
        // ---- B tile: 16 contiguous bf16 per thread (2x global b128),
        //      transposed scatter into BsT
        {
            const unsigned short* bp = &B[Bbase + (size_t)(kc * 32 + br) * NSP + nBase + bc];
            Pack8 t0, t1;
            t0.q = *(const uint4*)(bp);
            t1.q = *(const uint4*)(bp + 8);
            #pragma unroll
            for (int e = 0; e < 8; ++e) BsT[bc + e][br]     = t0.u[e];
            #pragma unroll
            for (int e = 0; e < 8; ++e) BsT[bc + 8 + e][br] = t1.u[e];
        }
        if (kc < 15) {
            __builtin_prefetch(&A[(size_t)(oBase + ar) * CDIM + (kc + 1) * 32 + ac], 0, 1);
            __builtin_prefetch(&B[Bbase + (size_t)((kc + 1) * 32 + br) * NSP + nBase + bc], 0, 1);
        }
        __syncthreads();

        v16bf af[4];
        #pragma unroll
        for (int mt = 0; mt < 4; ++mt)
            af[mt] = load_afrag(&As[wm * 64 + mt * 16 + ll][0], lh);
        v16bf bfg[2];
        #pragma unroll
        for (int nt = 0; nt < 2; ++nt)
            bfg[nt] = load_bfrag(&BsT[wn * 32 + nt * 16 + ll][0], lh);

        #pragma unroll
        for (int mt = 0; mt < 4; ++mt)
            #pragma unroll
            for (int nt = 0; nt < 2; ++nt)
                acc[mt * 2 + nt] = __builtin_amdgcn_wmma_f32_16x16x32_bf16(
                    false, af[mt], false, bfg[nt], (short)0, acc[mt * 2 + nt],
                    false, false);
        __syncthreads();
    }

    // epilogue
    #pragma unroll
    for (int mt = 0; mt < 4; ++mt) {
        #pragma unroll
        for (int nt = 0; nt < 2; ++nt) {
            #pragma unroll
            for (int r = 0; r < 8; ++r) {
                const int o = oBase + wm * 64 + mt * 16 + r + (lh << 3);
                const int n = nBase + wn * 32 + nt * 16 + ll;
                float val = acc[mt * 2 + nt][r] + bias[o];
                if (MODE == 0) {
                    if (o < 2 * CDIM) val *= qkscale;  // q and k rows
                    outb[((size_t)b * M + o) * NSP + n] = f2bf(val);
                } else {
                    val += resid[((size_t)b * CDIM + o) * NSP + n];
                    outf[((size_t)b * CDIM + o) * NSP + n] = val;
                }
            }
        }
    }
}

// ---------------------------------------------------------------------------
// Flash attention. qkv layout [b][o(1536)][n]: q rows [0,512), k [512,1024),
// v [1024,1536). 4 waves / WG (one per SIMD -> full register file per wave);
// each wave owns 16 query rows with Q fragments and O accumulators fully
// register-resident. Streams 64-wide key blocks; online softmax matched to
// the WMMA C/D row striping (lane half = row half). Output bf16 [b][c][n].
// ---------------------------------------------------------------------------
__global__ __launch_bounds__(128) __attribute__((amdgpu_waves_per_eu(1)))
void attn_kernel(const unsigned short* __restrict__ qkv,
                 unsigned short* __restrict__ outp)
{
    const int b  = blockIdx.y;
    const int i0 = blockIdx.x * 64;
    const int tid  = threadIdx.x;
    const int wave = tid >> 5;       // 0..3
    const int lane = tid & 31;
    const int lh   = lane >> 4;
    const int ll   = lane & 15;

    const size_t qb = (size_t)b * (3 * CDIM) * NSP;
    const unsigned short* q = qkv + qb;
    const unsigned short* k = qkv + qb + (size_t)CDIM * NSP;
    const unsigned short* v = qkv + qb + (size_t)(2 * CDIM) * NSP;

    __shared__ alignas(16) unsigned short KtT[64][40];    // [j][c] transposed
    __shared__ alignas(16) unsigned short Vt [64][72];    // [d][j] natural
    __shared__ alignas(16) unsigned short Pt [4][16][64]; // per-wave P tile

    const int irow = i0 + wave * 16;

    // cooperative-load coordinates (128 threads)
    const int kci = tid >> 2, kjj = (tid & 3) * 16;   // K chunk 32(c) x 64(j)
    const int vdd = tid >> 1, vjj = (tid & 1) * 32;   // V chunk 64(d) x 64(j)

    // Q fragments resident: 16 A-fragments covering c = 0..511 (one-time)
    v16bf qa[16];
    #pragma unroll
    for (int cc = 0; cc < 16; ++cc) {
        Frag16 f;
        #pragma unroll
        for (int i = 0; i < 16; ++i) {
            const int c = cc * 32 + ((i >> 3) << 4) + (lh << 3) + (i & 7);
            f.u[i] = q[(size_t)c * NSP + irow + ll];
        }
        qa[cc] = f.v;
    }

    v8f o[32];
    #pragma unroll
    for (int t = 0; t < 32; ++t)
        #pragma unroll
        for (int r = 0; r < 8; ++r) o[t][r] = 0.f;

    float mrow[8], lrow[8];
    #pragma unroll
    for (int r = 0; r < 8; ++r) { mrow[r] = -3.0e38f; lrow[r] = 0.f; }

    for (int jb = 0; jb < 64; ++jb) {
        const int j0 = jb * 64;

        // ---- S = Q K^T for this 16 x 64 row-block --------------------------
        v8f s[4];
        #pragma unroll
        for (int jt = 0; jt < 4; ++jt)
            #pragma unroll
            for (int r = 0; r < 8; ++r) s[jt][r] = 0.f;

        #pragma unroll
        for (int cc = 0; cc < 16; ++cc) {
            // K chunk: 16 contiguous bf16 per thread (2x global b128),
            // transposed scatter into KtT[j][c]
            {
                const unsigned short* kp = &k[(size_t)(cc * 32 + kci) * NSP + j0 + kjj];
                Pack8 k0, k1;
                k0.q = *(const uint4*)(kp);
                k1.q = *(const uint4*)(kp + 8);
                #pragma unroll
                for (int e = 0; e < 8; ++e) KtT[kjj + e][kci]     = k0.u[e];
                #pragma unroll
                for (int e = 0; e < 8; ++e) KtT[kjj + 8 + e][kci] = k1.u[e];
            }
            if (cc < 15)
                __builtin_prefetch(&k[(size_t)((cc + 1) * 32 + kci) * NSP + j0 + kjj], 0, 1);
            __syncthreads();
            #pragma unroll
            for (int jt = 0; jt < 4; ++jt) {
                v16bf bf = load_bfrag(&KtT[jt * 16 + ll][0], lh);
                s[jt] = __builtin_amdgcn_wmma_f32_16x16x32_bf16(
                    false, qa[cc], false, bf, (short)0, s[jt], false, false);
            }
            __syncthreads();
        }

        // ---- online softmax ------------------------------------------------
        float bm[8];
        #pragma unroll
        for (int r = 0; r < 8; ++r) {
            float mv = fmaxf(fmaxf(s[0][r], s[1][r]), fmaxf(s[2][r], s[3][r]));
            #pragma unroll
            for (int m = 1; m < 16; m <<= 1) mv = fmaxf(mv, __shfl_xor(mv, m, 16));
            bm[r] = mv;
        }
        float mnew[8], alpha[8], lsum[8];
        #pragma unroll
        for (int r = 0; r < 8; ++r) {
            mnew[r]  = fmaxf(mrow[r], bm[r]);
            alpha[r] = __expf(mrow[r] - mnew[r]);
            lsum[r]  = 0.f;
        }
        #pragma unroll
        for (int jt = 0; jt < 4; ++jt)
            #pragma unroll
            for (int r = 0; r < 8; ++r) {
                const float p = __expf(s[jt][r] - mnew[r]);
                float ps = p;
                #pragma unroll
                for (int m = 1; m < 16; m <<= 1) ps += __shfl_xor(ps, m, 16);
                lsum[r] += ps;
                Pt[wave][r + (lh << 3)][jt * 16 + ll] = f2bf(p);
            }
        #pragma unroll
        for (int r = 0; r < 8; ++r) {
            lrow[r] = lrow[r] * alpha[r] + lsum[r];
            mrow[r] = mnew[r];
        }
        #pragma unroll
        for (int t = 0; t < 32; ++t)
            #pragma unroll
            for (int r = 0; r < 8; ++r) o[t][r] *= alpha[r];
        __syncthreads();

        // ---- O += P V, streaming V in 64-wide d chunks ---------------------
        #pragma unroll
        for (int dc = 0; dc < 8; ++dc) {
            // V chunk: fully vectorized copy (4x b128 in, 4x b128 to LDS)
            {
                const unsigned short* vp = &v[(size_t)(dc * 64 + vdd) * NSP + j0 + vjj];
                uint4 a0 = *(const uint4*)(vp);
                uint4 a1 = *(const uint4*)(vp + 8);
                uint4 a2 = *(const uint4*)(vp + 16);
                uint4 a3 = *(const uint4*)(vp + 24);
                *(uint4*)&Vt[vdd][vjj]      = a0;
                *(uint4*)&Vt[vdd][vjj + 8]  = a1;
                *(uint4*)&Vt[vdd][vjj + 16] = a2;
                *(uint4*)&Vt[vdd][vjj + 24] = a3;
            }
            if (dc < 7)
                __builtin_prefetch(&v[(size_t)((dc + 1) * 64 + vdd) * NSP + j0 + vjj], 0, 1);
            __syncthreads();
            #pragma unroll
            for (int jc = 0; jc < 2; ++jc) {
                // P as A-fragment: runs K = jc*32 + 8h..+7 and +16
                Frag16 pf;
                const unsigned short* prow = &Pt[wave][ll][0];
                pf.q[0] = *(const uint4*)(prow + jc * 32 + 8 * lh);
                pf.q[1] = *(const uint4*)(prow + jc * 32 + 16 + 8 * lh);
                #pragma unroll
                for (int nt = 0; nt < 4; ++nt) {
                    Frag16 vf;
                    const unsigned short* vrow = &Vt[nt * 16 + ll][0];
                    vf.q[0] = *(const uint4*)(vrow + jc * 32 + 8 * lh);
                    vf.q[1] = *(const uint4*)(vrow + jc * 32 + 16 + 8 * lh);
                    o[dc * 4 + nt] = __builtin_amdgcn_wmma_f32_16x16x32_bf16(
                        false, pf.v, false, vf.v, (short)0, o[dc * 4 + nt],
                        false, false);
                }
            }
            __syncthreads();
        }
    }

    // ---- epilogue: O * (1/l), packed b128 stores, bf16 [b][d][i] -----------
    float rl[8];
    #pragma unroll
    for (int r = 0; r < 8; ++r) rl[r] = 1.0f / lrow[r];

    const size_t ob = (size_t)b * CDIM * NSP;
    #pragma unroll
    for (int t = 0; t < 32; ++t) {
        const int dc = t >> 2, nt = t & 3;
        const int d = dc * 64 + nt * 16 + ll;
        Pack8 pk;
        #pragma unroll
        for (int r = 0; r < 8; ++r) pk.u[r] = f2bf(o[t][r] * rl[r]);
        // rows r=0..7 map to contiguous i = irow + 8*lh + r
        *(uint4*)&outp[ob + (size_t)d * NSP + irow + 8 * lh] = pk.q;
    }
}

// ---------------------------------------------------------------------------
extern "C" void kernel_launch(void* const* d_in, const int* in_sizes, int n_in,
                              void* d_out, int out_size, void* d_ws, size_t ws_size,
                              hipStream_t stream)
{
    (void)in_sizes; (void)n_in; (void)out_size; (void)ws_size;
    const float* x      = (const float*)d_in[0];
    const float* gamma  = (const float*)d_in[1];
    const float* beta   = (const float*)d_in[2];
    const float* w_qkv  = (const float*)d_in[3];
    const float* b_qkv  = (const float*)d_in[4];
    const float* w_proj = (const float*)d_in[5];
    const float* b_proj = (const float*)d_in[6];

    unsigned short* xn    = (unsigned short*)d_ws;                    // 4*512*4096
    unsigned short* qkv   = xn    + (size_t)BATCH * CDIM * NSP;       // 4*1536*4096
    unsigned short* attno = qkv   + (size_t)BATCH * 3 * CDIM * NSP;   // 4*512*4096
    unsigned short* wqb   = attno + (size_t)BATCH * CDIM * NSP;       // 1536*512
    unsigned short* wpb   = wqb   + (size_t)3 * CDIM * CDIM;          // 512*512

    const float scale = 0.21022410381342864f;  // 512^-0.25 (applied to q and k)

    // weight conversion (one-shot, bandwidth-trivial)
    {
        const int n4q = (3 * CDIM * CDIM) / 4;   // 196608 float4s
        const int n4p = (CDIM * CDIM) / 4;       //  65536 float4s
        wconv_kernel<<<dim3((n4q + 255) / 256), dim3(256), 0, stream>>>(w_qkv, wqb, n4q);
        wconv_kernel<<<dim3((n4p + 255) / 256), dim3(256), 0, stream>>>(w_proj, wpb, n4p);
    }

    gn_kernel<<<dim3(BATCH * 32), dim3(256), 0, stream>>>(x, gamma, beta, xn);

    gemm_kernel<0><<<dim3(NSP / 128, (3 * CDIM) / 128, BATCH), dim3(256), 0, stream>>>(
        wqb, xn, b_qkv, qkv, nullptr, nullptr, 3 * CDIM, scale);

    attn_kernel<<<dim3(NSP / 64, BATCH), dim3(128), 0, stream>>>(qkv, attno);

    gemm_kernel<1><<<dim3(NSP / 128, CDIM / 128, BATCH), dim3(256), 0, stream>>>(
        wpb, attno, b_proj, nullptr, (float*)d_out, x, CDIM, 1.0f);
}
